// FlashMultiHeadAttention_35175782154739
// MI455X (gfx1250) — matrix operand order repaired and checked by reading, not compiled, not run on hardware
//
#include <hip/hip_runtime.h>
#include <hip/hip_bf16.h>

#define B_   2
#define S_   2048
#define HID_ 1024
#define H_   16
#define D_   64
#define NEG_ (-1000000000.0f)

typedef __attribute__((ext_vector_type(16))) __bf16 v16bf;
typedef __attribute__((ext_vector_type(8)))  __bf16 v8bf;
typedef __attribute__((ext_vector_type(8)))  float  v8f;

__device__ __forceinline__ __bf16 to_bf(float f) {
  unsigned u = __builtin_bit_cast(unsigned, f);
  u += 0x7fffu + ((u >> 16) & 1u);           // round-to-nearest-even
  unsigned short hs = (unsigned short)(u >> 16);
  return __builtin_bit_cast(__bf16, hs);
}

#define CAT16(lo, hi) __builtin_shufflevector(lo, hi, 0, 1, 2, 3, 4, 5, 6, 7, 8, 9, 10, 11, 12, 13, 14, 15)

// ---- CDNA5 async global->LDS copy path (ASYNCcnt) -------------------------
// LDS aperture flat addresses carry the LDS byte offset in addr[31:0]
// (ISA 10.2 aperture table), so the low dword of a generic pointer to a
// __shared__ object is the LDS address the async ops want.
__device__ __forceinline__ unsigned lds_addr32(const void* p) {
  return (unsigned)(unsigned long long)p;
}
__device__ __forceinline__ void async_b128(unsigned lds_dst, const void* src) {
  asm volatile("global_load_async_to_lds_b128 %0, %1, off"
               :: "v"(lds_dst), "v"(src) : "memory");
}
__device__ __forceinline__ void async_b32(unsigned lds_dst, const void* src) {
  asm volatile("global_load_async_to_lds_b32 %0, %1, off"
               :: "v"(lds_dst), "v"(src) : "memory");
}
__device__ __forceinline__ void wait_async0() {
  asm volatile("s_wait_asynccnt 0x0" ::: "memory");
}

// ---------------------------------------------------------------------------
// elementwise fp32 -> bf16 (x4 per thread)
// ---------------------------------------------------------------------------
__global__ __launch_bounds__(256) void cvt_f32_bf16(
    const float* __restrict__ s, __bf16* __restrict__ d, int n)
{
  const int i = (blockIdx.x * 256 + threadIdx.x) * 4;
  if (i + 3 < n) {
    const float4 f = *(const float4*)(s + i);
    d[i + 0] = to_bf(f.x);
    d[i + 1] = to_bf(f.y);
    d[i + 2] = to_bf(f.z);
    d[i + 3] = to_bf(f.w);
  }
}

// ---------------------------------------------------------------------------
// C[M,N] = A[M,K](bf16) @ W[N,K](bf16)^T + bias   (fp32 accumulate)
// Wave tile 16x64, block = 4 waves stacked in M -> 64x64.
// A frag (16x32): lane row = lane%16, elem e -> K = half*16+e (contiguous 32B)
// B frag (32x16): lane col = lane%16, elem e -> K = half*16+e (contiguous 32B)
// C tile: VGPR r, lane L -> (M = r + 8*(L/16), N = L%16)
// ---------------------------------------------------------------------------
__global__ __launch_bounds__(128) void gemm_bf16_xwT_bias(
    const __bf16* __restrict__ A, const __bf16* __restrict__ W,
    const float* __restrict__ bias, float* __restrict__ C,
    int N, int K)
{
  const int lane = threadIdx.x & 31;
  const int wave = threadIdx.x >> 5;
  const int half = (lane >> 4) & 1;
  const int ln   = lane & 15;
  const int b8   = half * 8;
  const int h16  = half * 16;

  const int nbase = blockIdx.x * 64;
  const int mbase = blockIdx.y * 64 + wave * 16;

  const __bf16* arow = A + (size_t)(mbase + ln) * K;

  v8f acc[4] = {};

  for (int k0 = 0; k0 < K; k0 += 32) {
    const v16bf af = *(const v16bf*)(arow + k0 + h16);
#pragma unroll
    for (int t = 0; t < 4; ++t) {
      const __bf16* wrow = W + (size_t)(nbase + t * 16 + ln) * K + k0;
      const v16bf bfr = *(const v16bf*)(wrow + h16);
      acc[t] = __builtin_amdgcn_wmma_f32_16x16x32_bf16(
          false, af, false, bfr, (short)0, acc[t], false, false);
    }
  }

#pragma unroll
  for (int t = 0; t < 4; ++t) {
    const int n = nbase + t * 16 + ln;
    const float bv = bias[n];
#pragma unroll
    for (int r = 0; r < 8; ++r) {
      const int m = mbase + r + 8 * half;
      C[(size_t)m * N + n] = acc[t][r] + bv;
    }
  }
}

// ---------------------------------------------------------------------------
// RoPE (reference-exact) + bf16 conversion.
// Q,K -> [B,H,S,D]; V -> transposed [B,H,D,S] for contiguous async staging.
// ---------------------------------------------------------------------------
__global__ __launch_bounds__(64) void rope_convert(
    const float* __restrict__ Qf, const float* __restrict__ Kf,
    const float* __restrict__ Vf,
    __bf16* __restrict__ Qbf, __bf16* __restrict__ Kbf, __bf16* __restrict__ VbfT)
{
  const int j = threadIdx.x;               // 0..63 within head dim
  const int idx = blockIdx.x;              // b*S*H + s*H + h
  const int h = idx % H_;
  const int s = (idx / H_) % S_;
  const int b = idx / (H_ * S_);
  const size_t src = ((size_t)b * S_ + s) * HID_ + (size_t)h * D_;

  const int jm = j & 31;
  const float inv_freq = __expf(-((float)(2 * jm) / (float)D_) * 9.210340371976184f);
  const float ang = (float)s * inv_freq;
  const float c  = __cosf(ang);
  const float sn = __sinf(ang);

  const int   rotidx = (j < 32) ? (2 * j + 1) : (2 * (j - 32));
  const float rsign  = (j < 32) ? -1.0f : 1.0f;

  const float q = Qf[src + j] * c + rsign * Qf[src + rotidx] * sn;
  const float k = Kf[src + j] * c + rsign * Kf[src + rotidx] * sn;

  const size_t dst = (((size_t)b * H_ + h) * S_ + s) * D_ + j;
  Qbf[dst] = to_bf(q);
  Kbf[dst] = to_bf(k);
  VbfT[(((size_t)b * H_ + h) * D_ + j) * S_ + s] = to_bf(Vf[src + j]);
}

// ---------------------------------------------------------------------------
// Flash attention, causal + relative-position bias.
// 64 q-rows/block (4 waves x 16). 32-key tiles double-buffered in LDS via
// global_load_async_to_lds_b128; rel_emb band staged per tile via async b32.
// ---------------------------------------------------------------------------
__global__ __launch_bounds__(128) void flash_attn(
    const __bf16* __restrict__ Qbf, const __bf16* __restrict__ Kbf,
    const __bf16* __restrict__ VbfT, const float* __restrict__ rel_emb,
    __bf16* __restrict__ Obf)
{
  __shared__ __bf16 ldsK[2][32 * 64];    // [key][d]
  __shared__ __bf16 ldsVt[2][64 * 32];   // [d][key]
  __shared__ float  ldsRel[2][96];       // rel_emb diagonal band
  __shared__ __bf16 ldsP[4][16 * 32];    // per-wave P staging (C->A layout)

  const int lane = threadIdx.x & 31;
  const int wave = threadIdx.x >> 5;
  const int half = (lane >> 4) & 1;
  const int ln   = lane & 15;
  const int b8   = half * 8;
  const int h16  = half * 16;

  const int qb    = blockIdx.x;
  const int h     = blockIdx.y;
  const int b     = blockIdx.z;
  const int qbase = qb * 64;
  const int qrow  = qbase + wave * 16;

  const __bf16* Qh  = Qbf  + ((size_t)b * H_ + h) * S_ * D_;
  const __bf16* Kh  = Kbf  + ((size_t)b * H_ + h) * S_ * D_;
  const __bf16* VTh = VbfT + ((size_t)b * H_ + h) * D_ * S_;
  const float*  relh = rel_emb + h;

  // Preload Q A-fragments (both 32-wide K-steps over D=64).
  v16bf qf[2];
  {
    const __bf16* qr = Qh + (size_t)(qrow + ln) * D_;
    qf[0] = *(const v16bf*)(qr + h16);
    qf[1] = *(const v16bf*)(qr + 32 + h16);
  }

  // async stage of one 32-key tile into buffer `buf`
  auto issue_tile = [&](int buf, int kb2) {
    const int kbase2 = kb2 * 32;
    // K tile: contiguous 4KB
    const char* gk = (const char*)(Kh + (size_t)kbase2 * D_);
    const unsigned lk = lds_addr32(&ldsK[buf][0]);
    for (int i = threadIdx.x; i < 256; i += 128)
      async_b128(lk + i * 16, gk + i * 16);
    // V^T tile: 64 rows x 64B, global row stride S_*2 bytes
    const unsigned lv = lds_addr32(&ldsVt[buf][0]);
    for (int i = threadIdx.x; i < 256; i += 128) {
      const int dd = i >> 2, ch = i & 3;
      async_b128(lv + dd * 64 + ch * 16,
                 (const char*)(VTh + (size_t)dd * S_ + kbase2) + ch * 16);
    }
    // rel_emb band: diffs [qbase-kbase-31, qbase-kbase+63], 95 values
    const int dmin = qbase - kbase2 - 31;
    const unsigned lr = lds_addr32(&ldsRel[buf][0]);
    if (threadIdx.x < 95)
      async_b32(lr + threadIdx.x * 4,
                relh + (size_t)(dmin + threadIdx.x + 2047) * H_);
  };

  v8f oacc[4] = {};
  float mrow[8], lrow[8];
#pragma unroll
  for (int r = 0; r < 8; ++r) { mrow[r] = -3.0e38f; lrow[r] = 0.0f; }

  const float scale = 0.125f;          // D^-0.5
  const int nkb = qb * 2 + 2;          // causal: keys up to qbase+63

  issue_tile(0, 0);
  wait_async0();
  __syncthreads();

  for (int kb = 0; kb < nkb; ++kb) {
    const int cur = kb & 1;
    const int kbase = kb * 32;

    if (kb + 1 < nkb) issue_tile(cur ^ 1, kb + 1);  // overlap next tile

    // ---- scores: Q (16xD) @ K^T -> 16x32 ----
    v8f sacc[2] = {};
#pragma unroll
    for (int t = 0; t < 2; ++t) {
      const __bf16* krow = &ldsK[cur][(t * 16 + ln) * 64];
#pragma unroll
      for (int s2 = 0; s2 < 2; ++s2) {
        const v16bf kfb = *(const v16bf*)(krow + s2 * 32 + h16);
        sacc[t] = __builtin_amdgcn_wmma_f32_16x16x32_bf16(
            false, qf[s2], false, kfb, (short)0, sacc[t], false, false);
      }
    }

    // ---- scale + rel-bias (LDS band, kb-invariant index) + causal mask ----
    float sc[2][8];
#pragma unroll
    for (int t = 0; t < 2; ++t) {
      const int kcol = kbase + t * 16 + ln;
#pragma unroll
      for (int r = 0; r < 8; ++r) {
        const int q = qrow + r + 8 * half;
        float v = sacc[t][r] * scale +
                  ldsRel[cur][wave * 16 + r + 8 * half + 31 - t * 16 - ln];
        if (kcol > q) v += NEG_;
        sc[t][r] = v;
      }
    }

    // ---- online softmax (rows live in 16-lane half-wave groups) ----
    float corr[8];
#pragma unroll
    for (int r = 0; r < 8; ++r) {
      float mx = fmaxf(sc[0][r], sc[1][r]);
#pragma unroll
      for (int off = 1; off < 16; off <<= 1)
        mx = fmaxf(mx, __shfl_xor(mx, off, 32));
      const float mnew = fmaxf(mrow[r], mx);
      corr[r] = __expf(mrow[r] - mnew);
      sc[0][r] = __expf(sc[0][r] - mnew);
      sc[1][r] = __expf(sc[1][r] - mnew);
      float rs = sc[0][r] + sc[1][r];
#pragma unroll
      for (int off = 1; off < 16; off <<= 1)
        rs += __shfl_xor(rs, off, 32);
      lrow[r] = lrow[r] * corr[r] + rs;
      mrow[r] = mnew;
    }
#pragma unroll
    for (int t = 0; t < 4; ++t)
#pragma unroll
      for (int r = 0; r < 8; ++r) oacc[t][r] *= corr[r];

    // ---- stage P (C layout) -> LDS -> A layout fragment ----
    __bf16* pw = &ldsP[wave][0];
#pragma unroll
    for (int t = 0; t < 2; ++t)
#pragma unroll
      for (int r = 0; r < 8; ++r)
        pw[(r + 8 * half) * 32 + t * 16 + ln] = to_bf(sc[t][r]);
    // wave-private region; same-wave LDS ops stay ordered.

    v16bf pf;
    {
      const __bf16* prow = &ldsP[wave][ln * 32];
      pf = *(const v16bf*)(prow + h16);
    }

    // ---- O += P (16x32) @ V (32x64) ----
#pragma unroll
    for (int t = 0; t < 4; ++t) {
      const v16bf vfb = *(const v16bf*)(&ldsVt[cur][(t * 16 + ln) * 32 + h16]);
      oacc[t] = __builtin_amdgcn_wmma_f32_16x16x32_bf16(
          false, pf, false, vfb, (short)0, oacc[t], false, false);
    }

    wait_async0();     // own async copies into buf cur^1 complete
    __syncthreads();   // whole block done with cur, next tile resident
  }

  // ---- epilogue: normalize, clip, store bf16 [B,S,HID] ----
#pragma unroll
  for (int t = 0; t < 4; ++t)
#pragma unroll
    for (int r = 0; r < 8; ++r) {
      float o = oacc[t][r] / lrow[r];
      o = fminf(30.0f, fmaxf(-30.0f, o));
      const int q = qrow + r + 8 * half;
      const int d = t * 16 + ln;
      Obf[((size_t)b * S_ + q) * HID_ + h * D_ + d] = to_bf(o);
    }
}

// ---------------------------------------------------------------------------
extern "C" void kernel_launch(void* const* d_in, const int* in_sizes, int n_in,
                              void* d_out, int out_size, void* d_ws, size_t ws_size,
                              hipStream_t stream) {
  (void)in_sizes; (void)n_in; (void)out_size; (void)ws_size;

  const float* query  = (const float*)d_in[0];
  const float* key_in = (const float*)d_in[1];
  const float* value  = (const float*)d_in[2];
  // d_in[3] = attn_mask (causal tril) -> implemented analytically
  const float* wq  = (const float*)d_in[4];
  const float* bq  = (const float*)d_in[5];
  const float* wk  = (const float*)d_in[6];
  const float* bk  = (const float*)d_in[7];
  const float* wv  = (const float*)d_in[8];
  const float* bv  = (const float*)d_in[9];
  const float* wo  = (const float*)d_in[10];
  const float* bo  = (const float*)d_in[11];
  const float* rel = (const float*)d_in[12];

  char* ws = (char*)d_ws;
  const size_t NF = (size_t)B_ * S_ * HID_;        // 4M activation elements
  const size_t WN = (size_t)HID_ * HID_;           // 1M weight elements
  float*  Qf   = (float*)(ws);                     // fp32 proj outputs (16MB ea)
  float*  Kf   = (float*)(ws + NF * 4);
  float*  Vf   = (float*)(ws + NF * 8);
  __bf16* Qbf  = (__bf16*)(ws + NF * 12);          // [B,H,S,D]
  __bf16* Kbf  = (__bf16*)(ws + NF * 14);
  __bf16* VbfT = (__bf16*)(ws + NF * 16);          // [B,H,D,S]
  __bf16* Obf  = (__bf16*)(ws + NF * 18);          // [B,S,HID]
  __bf16* Xq   = (__bf16*)(ws + NF * 20);          // bf16 activations
  __bf16* Xk   = (__bf16*)(ws + NF * 22);
  __bf16* Xv   = (__bf16*)(ws + NF * 24);
  __bf16* Wqb  = (__bf16*)(ws + NF * 26);          // bf16 weights (2MB ea)
  __bf16* Wkb  = Wqb + WN;
  __bf16* Wvb  = Wkb + WN;
  __bf16* Wob  = Wvb + WN;

  // one-time bf16 conversions (HBM-bound elementwise passes)
  cvt_f32_bf16<<<dim3((int)(NF / 1024)), 256, 0, stream>>>(query,  Xq,  (int)NF);
  cvt_f32_bf16<<<dim3((int)(NF / 1024)), 256, 0, stream>>>(key_in, Xk,  (int)NF);
  cvt_f32_bf16<<<dim3((int)(NF / 1024)), 256, 0, stream>>>(value,  Xv,  (int)NF);
  cvt_f32_bf16<<<dim3((int)(WN / 1024)), 256, 0, stream>>>(wq,     Wqb, (int)WN);
  cvt_f32_bf16<<<dim3((int)(WN / 1024)), 256, 0, stream>>>(wk,     Wkb, (int)WN);
  cvt_f32_bf16<<<dim3((int)(WN / 1024)), 256, 0, stream>>>(wv,     Wvb, (int)WN);
  cvt_f32_bf16<<<dim3((int)(WN / 1024)), 256, 0, stream>>>(wo,     Wob, (int)WN);

  const dim3 ggrid(HID_ / 64, (B_ * S_) / 64);     // (16, 64)
  gemm_bf16_xwT_bias<<<ggrid, 128, 0, stream>>>(Xq, Wqb, bq, Qf, HID_, HID_);
  gemm_bf16_xwT_bias<<<ggrid, 128, 0, stream>>>(Xk, Wkb, bk, Kf, HID_, HID_);
  gemm_bf16_xwT_bias<<<ggrid, 128, 0, stream>>>(Xv, Wvb, bv, Vf, HID_, HID_);

  rope_convert<<<dim3(B_ * S_ * H_), 64, 0, stream>>>(Qf, Kf, Vf, Qbf, Kbf, VbfT);

  flash_attn<<<dim3(S_ / 64, H_, B_), 128, 0, stream>>>(Qbf, Kbf, VbfT, rel, Obf);

  gemm_bf16_xwT_bias<<<ggrid, 128, 0, stream>>>(Obf, Wob, bo, (float*)d_out, HID_, HID_);
}
